// ComplexToReal_24584392802923
// MI455X (gfx1250) — compile-verified, matrix operand order checked by reading, not run on hardware
//
#include <hip/hip_runtime.h>
#include <math.h>

typedef __attribute__((ext_vector_type(8))) int   v8i;
typedef __attribute__((ext_vector_type(4))) float f4;

#define TLEN  3000
#define TVEC  (TLEN / 4)                     // 750 float4s per row (exact)
#define BLOCK 256
#define CHUNK ((TLEN + BLOCK - 1) / BLOCK)   // 12 elements per thread (scan phases)

__global__ __launch_bounds__(BLOCK)
void coherent_polar_kernel(const float* __restrict__ xr,
                           const float* __restrict__ xi,
                           float* __restrict__ out, int C)
{
    __shared__ float phase_s[TLEN];    // per-row phase, later unwrapped phase
    __shared__ int   corr_s[TLEN];     // intra-chunk inclusive step counts
    __shared__ int   sums_s[BLOCK];    // per-thread chunk step sums
    __shared__ int   scan_s[BLOCK];    // WMMA row-scan result
    __shared__ int   rowoff_s[16];     // cross-row exclusive offsets
    __shared__ float redmin_s[BLOCK];
    __shared__ float redmax_s[BLOCK];

    const int tid = threadIdx.x;
    const int row = blockIdx.x;                 // in [0, B*C)
    const int b   = row / C;
    const int c   = row % C;

    const size_t in_base  = (size_t)row * TLEN;
    const size_t mag_base = ((size_t)(b * 2 * C + c)) * TLEN;
    const size_t ph_base  = ((size_t)(b * 2 * C + C + c)) * TLEN;

    // ---- Pass A (coalesced, 128-bit): magnitude -> global NT, phase -> LDS -
    {
        const f4* xr4  = (const f4*)(xr + in_base);
        const f4* xi4  = (const f4*)(xi + in_base);
        f4*       mag4 = (f4*)(out + mag_base);
        for (int q = tid; q < TVEC; q += BLOCK) {
            f4 r  = __builtin_nontemporal_load(&xr4[q]);
            f4 im = __builtin_nontemporal_load(&xi4[q]);
            f4 m;
            #pragma unroll
            for (int j = 0; j < 4; ++j) {
                m[j] = sqrtf(r[j] * r[j] + im[j] * im[j]);
                phase_s[4 * q + j] = atan2f(im[j], r[j]);
            }
            __builtin_nontemporal_store(m, &mag4[q]);
        }
    }
    __syncthreads();

    // ---- Pass B (chunked, LDS-only): integer wrap-step counts --------------
    const int beg = tid * CHUNK;
    const int end = min(beg + CHUNK, TLEN);
    const float PI = 3.14159265358979323846f;
    int run = 0;
    for (int idx = beg; idx < end; ++idx) {
        if (idx > 0) {
            float d = phase_s[idx] - phase_s[idx - 1];
            run += (d < -PI) ? 1 : 0;
            run -= (d >  PI) ? 1 : 0;
        }
        corr_s[idx] = run;                      // inclusive within chunk
    }
    sums_s[tid] = run;                          // |run| <= CHUNK = 12, fits i8
    __syncthreads();

    // ---- Pass C: 256-wide scan; within-row (16x16) scans via one
    //      V_WMMA_I32_16X16X64_IU8: D = X (signed i8, K zero-padded) x U ----
    if (tid < 32) {
        const int lane = tid;
        const int half = lane >> 4;             // lane group
        const int lm   = lane & 15;             // A: row M; B/D: column N
        const int kb   = half * 8;              // this group's K range

        v8i A  = (v8i){0, 0, 0, 0, 0, 0, 0, 0};
        v8i Bm = (v8i){0, 0, 0, 0, 0, 0, 0, 0};
        unsigned a0p = 0, a1p = 0, b0p = 0, b1p = 0;
        #pragma unroll
        for (int byt = 0; byt < 4; ++byt) {
            int k0 = kb + byt;
            int k1 = kb + 4 + byt;
            // A layout: V0 bytes = K kb..kb+3, V1 bytes = K kb+4..kb+7 (row lm)
            a0p |= ((unsigned)(sums_s[lm * 16 + k0] & 0xff)) << (8 * byt);
            a1p |= ((unsigned)(sums_s[lm * 16 + k1] & 0xff)) << (8 * byt);
            // B = upper-triangular ones: B[k][n] = (k <= n), rows K>=16 are 0
            b0p |= ((k0 <= lm) ? 1u : 0u) << (8 * byt);
            b1p |= ((k1 <= lm) ? 1u : 0u) << (8 * byt);
        }
        A[0]  = (int)a0p;  A[1]  = (int)a1p;    // K = 16..63 zero-padded
        Bm[0] = (int)b0p;  Bm[1] = (int)b1p;

        v8i Cz = (v8i){0, 0, 0, 0, 0, 0, 0, 0};
        v8i D  = __builtin_amdgcn_wmma_i32_16x16x64_iu8(
            /*sgn_a=*/true, A, /*sgn_b=*/false, Bm, Cz,
            /*reuse_a=*/false, /*reuse_b=*/false);

        // D layout: lane holds column N=lm; VGPR v -> row M = v + 8*half
        #pragma unroll
        for (int v = 0; v < 8; ++v)
            scan_s[(v + 8 * half) * 16 + lm] = D[v];
    }
    __syncthreads();
    if (tid == 0) {                             // 16-wide cross-row fixup
        int acc = 0;
        #pragma unroll
        for (int r = 0; r < 16; ++r) { rowoff_s[r] = acc; acc += scan_s[r * 16 + 15]; }
    }
    __syncthreads();

    // exclusive prefix of wrap counts entering this thread's chunk
    const int base = scan_s[tid] + rowoff_s[tid >> 4] - run;

    // ---- Pass D (chunked, LDS-only): unwrap, track local min/max -----------
    const float TWO_PI = 6.28318530717958647692f;
    float mn = INFINITY, mx = -INFINITY;
    for (int idx = beg; idx < end; ++idx) {
        float u = phase_s[idx] + TWO_PI * (float)(base + corr_s[idx]);
        phase_s[idx] = u;
        mn = fminf(mn, u);
        mx = fmaxf(mx, u);
    }
    redmin_s[tid] = mn;
    redmax_s[tid] = mx;
    __syncthreads();
    for (int s = BLOCK / 2; s > 0; s >>= 1) {
        if (tid < s) {
            redmin_s[tid] = fminf(redmin_s[tid], redmin_s[tid + s]);
            redmax_s[tid] = fmaxf(redmax_s[tid], redmax_s[tid + s]);
        }
        __syncthreads();
    }
    const float gmn = redmin_s[0];
    const float inv = 1.0f / (redmax_s[0] - gmn + 1e-8f);

    // ---- Pass E (coalesced, 128-bit): normalize, NT stream out -------------
    {
        f4* ph4 = (f4*)(out + ph_base);
        for (int q = tid; q < TVEC; q += BLOCK) {
            f4 u;
            #pragma unroll
            for (int j = 0; j < 4; ++j)
                u[j] = (phase_s[4 * q + j] - gmn) * inv;
            __builtin_nontemporal_store(u, &ph4[q]);
        }
    }
}

extern "C" void kernel_launch(void* const* d_in, const int* in_sizes, int n_in,
                              void* d_out, int out_size, void* d_ws, size_t ws_size,
                              hipStream_t stream) {
    const float* xr = (const float*)d_in[0];
    const float* xi = (const float*)d_in[1];
    float* out = (float*)d_out;

    const int C    = 1025;                 // reference: B=8, C=1025, T=3000
    const int rows = in_sizes[0] / TLEN;   // B*C = 8200 workgroups

    coherent_polar_kernel<<<rows, BLOCK, 0, stream>>>(xr, xi, out, C);

    (void)n_in; (void)out_size; (void)d_ws; (void)ws_size;
}
